// MultiHeadAttention_31988916420894
// MI455X (gfx1250) — matrix operand order repaired and checked by reading, not compiled, not run on hardware
//
#include <hip/hip_runtime.h>
#include <math.h>

// ---------------------------------------------------------------------------
// MHA forward for MI455X (gfx1250, wave32, WMMA bf16 path + async-to-LDS)
// B=2, S=2048, D=1024, H=16, DK=64
// ---------------------------------------------------------------------------

#define Bc 2
#define Sc 2048
#define Dc 1024
#define Hc 16
#define DKc 64
#define Mc (Bc * Sc)          // 4096 rows in all GEMMs

typedef __bf16 bf16_t;
typedef __bf16 bf16x4  __attribute__((ext_vector_type(4)));
typedef __bf16 bf16x8  __attribute__((ext_vector_type(8)));
typedef __bf16 v16bf   __attribute__((ext_vector_type(16)));
typedef float  v8f     __attribute__((ext_vector_type(8)));

#define WMMA_BF16(a, b, c) \
  __builtin_amdgcn_wmma_f32_16x16x32_bf16(false, (a), false, (b), (short)0, (c), false, false)

// A-matrix (16x32 bf16) fragment: lane half h holds K = {h*8..h*8+7, 16+h*8..16+h*8+7}
__device__ __forceinline__ v16bf load_a_frag(const bf16_t* rowp, int half) {
  bf16x8 lo = *(const bf16x8*)(rowp + half * 8);
  bf16x8 hi = *(const bf16x8*)(rowp + 16 + half * 8);
  v16bf f;
#pragma unroll
  for (int e = 0; e < 8; ++e) { f[e] = lo[e]; f[e + 8] = hi[e]; }
  return f;
}

// B-matrix (32x16 bf16) fragment: lane half h holds K = {h*8..h*8+7, 16+h*8..16+h*8+7}
__device__ __forceinline__ v16bf load_b_frag(const bf16_t* rowp, int half) {
  bf16x8 lo = *(const bf16x8*)(rowp + half * 8);
  bf16x8 hi = *(const bf16x8*)(rowp + 16 + half * 8);
  v16bf f;
#pragma unroll
  for (int e = 0; e < 8; ++e) { f[e] = lo[e]; f[e + 8] = hi[e]; }
  return f;
}

// CDNA5 async global->LDS copy (16B/lane). INST_OFFSET is added to BOTH the
// LDS address (VDST VGPR) and the global address (VADDR), so one immediate
// advances both sides — no per-chunk VALU address math.
template <int OFS>
__device__ __forceinline__ void async_ld_b128(unsigned lds_off, const void* gaddr) {
  asm volatile("global_load_async_to_lds_b128 %0, %1, off offset:%c2"
               :: "v"(lds_off), "v"(gaddr), "n"(OFS)
               : "memory");
}
template <int N>
__device__ __forceinline__ void wait_asynccnt() {
  asm volatile("s_wait_asynccnt %c0" :: "n"(N) : "memory");
}

// ---------------------------------------------------------------------------
// Kernel 0: fp32 -> bf16 cast (vector x4)
// ---------------------------------------------------------------------------
__global__ void cast_f32_bf16(const float* __restrict__ src,
                              bf16_t* __restrict__ dst, int n) {
  int i = (blockIdx.x * blockDim.x + threadIdx.x) * 4;
  if (i + 3 < n) {
    float4 v = *(const float4*)(src + i);
    bf16x4 o;
    o[0] = (bf16_t)v.x; o[1] = (bf16_t)v.y; o[2] = (bf16_t)v.z; o[3] = (bf16_t)v.w;
    *(bf16x4*)(dst + i) = o;
  }
}

// ---------------------------------------------------------------------------
// Kernel 1: GEMM  Y = X @ W^T + bias   (M=4096, N=1024, K=1024, bf16->f32 acc)
// 128x128 workgroup tile, 8 waves (4x2), each wave owns 2x4 16x16 tiles.
// K pipelined in 64-wide stages, double-buffered via async-to-LDS:
// per stage each thread stages one 128B row slice (8 async b128),
// 2 barriers cover 16 WMMAs.
// mode 0: bf16 head-major [B,H,S,DK]
// mode 1: fp32 flat [M,N]
// mode 2: bf16 head-major TRANSPOSED [B,H,DK,S]   (for V)
// ---------------------------------------------------------------------------
__launch_bounds__(256)
__global__ void gemm_bf16(const bf16_t* __restrict__ X,
                          const bf16_t* __restrict__ W,
                          const float*  __restrict__ bias,
                          void* __restrict__ Yout, int mode) {
  // per stage: A = 128 rows x 64 K (16 KB), B = 128 rows x 64 K (16 KB)
  __shared__ __align__(64) bf16_t ABuf[2][16384];   // 64 KB double buffered

  const int m0 = blockIdx.x * 128;
  const int n0 = blockIdx.y * 128;
  const int t    = threadIdx.x;
  const int lane = t & 31, wave = t >> 5;
  const int half = lane >> 4, r = lane & 15;
  const int wy = wave >> 1;            // 0..3 : M sub-block (32 rows)
  const int wx = wave & 1;             // 0..1 : N sub-block (64 cols)

  v8f acc[2][4] = {};

  // scalar-uniform branch: waves 0-3 stage A rows, waves 4-7 stage B rows
  auto issue_stage = [&](int stg, int k0) {
    const unsigned lbase = (unsigned)(uintptr_t)(&ABuf[stg][0]);
    const int wgrp = __builtin_amdgcn_readfirstlane(wave);
    if (wgrp < 4) {
      const int row = wave * 32 + lane;            // A rows 0..127
      const bf16_t* g = &X[(size_t)(m0 + row) * Dc + k0];
      const unsigned l = lbase + (unsigned)(row * 128);
      async_ld_b128<0>(l, g);   async_ld_b128<16>(l, g);
      async_ld_b128<32>(l, g);  async_ld_b128<48>(l, g);
      async_ld_b128<64>(l, g);  async_ld_b128<80>(l, g);
      async_ld_b128<96>(l, g);  async_ld_b128<112>(l, g);
    } else {
      const int row = (wave - 4) * 32 + lane;      // B rows 0..127
      const bf16_t* g = &W[(size_t)(n0 + row) * Dc + k0];
      const unsigned l = lbase + 32768u + (unsigned)(row * 128);
      async_ld_b128<0>(l, g);   async_ld_b128<16>(l, g);
      async_ld_b128<32>(l, g);  async_ld_b128<48>(l, g);
      async_ld_b128<64>(l, g);  async_ld_b128<80>(l, g);
      async_ld_b128<96>(l, g);  async_ld_b128<112>(l, g);
    }
  };

  const int NK = Dc / 64;              // 16 pipeline stages
  issue_stage(0, 0);
  for (int ks = 0; ks < NK; ++ks) {
    if (ks + 1 < NK) {
      issue_stage((ks + 1) & 1, (ks + 1) * 64);
      wait_asynccnt<8>();              // stage ks landed (8 newest remain)
    } else {
      wait_asynccnt<0>();
    }
    __syncthreads();                   // all waves' stage-ks data visible

    const bf16_t* Ab = &ABuf[ks & 1][0];
    const bf16_t* Bb = &ABuf[ks & 1][16384 / 2];
#pragma unroll
    for (int slab = 0; slab < 2; ++slab) {         // two 32-wide K slabs
      v16bf af[2], bfr[4];
#pragma unroll
      for (int mi = 0; mi < 2; ++mi)
        af[mi] = load_a_frag(&Ab[(wy * 32 + mi * 16 + r) * 64 + slab * 32], half);
#pragma unroll
      for (int ni = 0; ni < 4; ++ni)
        bfr[ni] = load_b_frag(&Bb[(wx * 64 + ni * 16 + r) * 64 + slab * 32], half);
#pragma unroll
      for (int mi = 0; mi < 2; ++mi)
#pragma unroll
        for (int ni = 0; ni < 4; ++ni)
          acc[mi][ni] = WMMA_BF16(af[mi], bfr[ni], acc[mi][ni]);
    }
    __syncthreads();                   // safe to overwrite buffer ks&1 later
  }

#pragma unroll
  for (int mi = 0; mi < 2; ++mi)
#pragma unroll
    for (int ni = 0; ni < 4; ++ni)
#pragma unroll
      for (int i = 0; i < 8; ++i) {
        const int gm = m0 + wy * 32 + mi * 16 + half * 8 + i;
        const int gn = n0 + wx * 64 + ni * 16 + r;
        const float y = acc[mi][ni][i] + bias[gn];
        if (mode == 1) {
          ((float*)Yout)[(size_t)gm * Dc + gn] = y;
        } else {
          bf16_t* Y = (bf16_t*)Yout;
          const int b_ = gm >> 11, s_ = gm & (Sc - 1);
          const int h_ = gn >> 6,  d_ = gn & 63;
          if (mode == 0)
            Y[(((size_t)(b_ * Hc + h_)) * Sc + s_) * DKc + d_] = (bf16_t)y;
          else  // mode 2: transposed per head [B,H,DK,S]
            Y[(((size_t)(b_ * Hc + h_)) * DKc + d_) * Sc + s_] = (bf16_t)y;
        }
      }
}

// ---------------------------------------------------------------------------
// Kernel 2: fused attention for one (b,h, 16-query strip)
//   scores = scale * Q K^T  -> LDS (16x2048 fp32, 128 KB)
//   softmax in LDS, write attn once (fp32)
//   out = P @ V via WMMA; V is pre-transposed so B-fragments are contiguous
// grid: (S/16, B*H), block: 256 (8 waves)
// ---------------------------------------------------------------------------
__launch_bounds__(256)
__global__ void attn_fused(const bf16_t* __restrict__ Qh,
                           const bf16_t* __restrict__ Kh,
                           const bf16_t* __restrict__ Vt,   // [B,H,DK,S]
                           float* __restrict__ attn,
                           bf16_t* __restrict__ Obuf) {
  __shared__ float  Ps[16 * Sc];        // 128 KB score strip
  __shared__ __align__(16) bf16_t Qs[16 * DKc];   // 2 KB Q strip
  __shared__ float  part[8][16 * DKc];  // 32 KB P@V partials per wave
  __shared__ float  redbuf[16 * 16];
  __shared__ float  rowmax[16];
  __shared__ float  rowinv[16];

  const int qs = blockIdx.x * 16;
  const int bh = blockIdx.y;
  const int b_ = bh >> 4, h_ = bh & 15;
  const bf16_t* Qb  = Qh + (size_t)bh * Sc * DKc;
  const bf16_t* Kb  = Kh + (size_t)bh * Sc * DKc;
  const bf16_t* Vtb = Vt + (size_t)bh * DKc * Sc;

  const int t    = threadIdx.x;
  const int lane = t & 31, wave = t >> 5;
  const int half = lane >> 4, r = lane & 15;

  // ---- stage Q strip (16 x 64 bf16) ----
  if (t < 128) {
    const int qr = t >> 3, qc = (t & 7) * 8;
    *(bf16x8*)&Qs[qr * DKc + qc] =
        *(const bf16x8*)&Qb[(size_t)(qs + qr) * DKc + qc];
  }
  __syncthreads();

  // ---- phase 1: scores = scale * Q K^T ; wave w owns key cols [w*256,+256) ----
  const float scale = 0.125f;  // 1/sqrt(64)
  v16bf a0 = load_a_frag(&Qs[r * DKc + 0], half);
  v16bf a1 = load_a_frag(&Qs[r * DKc + 32], half);
  for (int tt = 0; tt < 16; ++tt) {
    const int n0 = wave * 256 + tt * 16;
    const bf16_t* krow = Kb + (size_t)(n0 + r) * DKc;
    v16bf b0 = load_b_frag(krow, half);
    v16bf b1 = load_b_frag(krow + 32, half);
    v8f acc = {};
    acc = WMMA_BF16(a0, b0, acc);
    acc = WMMA_BF16(a1, b1, acc);
#pragma unroll
    for (int i = 0; i < 8; ++i)
      Ps[(half * 8 + i) * Sc + n0 + r] = acc[i] * scale;
  }
  __syncthreads();

  // ---- phase 2: softmax over each of the 16 rows (2048 wide) ----
  const int srow = t >> 4, scol = t & 15;
  float mx = -INFINITY;
  for (int c = scol; c < Sc; c += 16) mx = fmaxf(mx, Ps[srow * Sc + c]);
  redbuf[srow * 16 + scol] = mx;
  __syncthreads();
  if (t < 16) {
    float mm = redbuf[t * 16];
#pragma unroll
    for (int j = 1; j < 16; ++j) mm = fmaxf(mm, redbuf[t * 16 + j]);
    rowmax[t] = mm;
  }
  __syncthreads();
  const float rm = rowmax[srow];
  float ssum = 0.f;
  for (int c = scol; c < Sc; c += 16) {
    float e = __expf(Ps[srow * Sc + c] - rm);
    Ps[srow * Sc + c] = e;
    ssum += e;
  }
  redbuf[srow * 16 + scol] = ssum;
  __syncthreads();
  if (t < 16) {
    float s = 0.f;
#pragma unroll
    for (int j = 0; j < 16; ++j) s += redbuf[t * 16 + j];
    rowinv[t] = 1.0f / s;
  }
  __syncthreads();

  // ---- normalize in LDS and write attn strip (single mandatory write) ----
  float* attn_base = attn + ((size_t)bh * Sc + qs) * Sc;   // 16 contiguous rows
#pragma unroll
  for (int i = 0; i < 32; ++i) {
    const int idx = t * 4 + i * 1024;
    const float inv = rowinv[idx >> 11];
    float4 v4 = *(float4*)&Ps[idx];
    v4.x *= inv; v4.y *= inv; v4.z *= inv; v4.w *= inv;
    *(float4*)&Ps[idx] = v4;
    *(float4*)&attn_base[idx] = v4;
  }
  __syncthreads();

  // ---- phase 3: out = P @ V ; wave w reduces K range [w*256,+256) ----
  v8f accv[4] = {};
  for (int kk = wave * 256; kk < wave * 256 + 256; kk += 32) {
    v16bf a;
#pragma unroll
    for (int e = 0; e < 8; ++e) {
      a[e]     = (bf16_t)Ps[r * Sc + kk + half * 8 + e];
      a[e + 8] = (bf16_t)Ps[r * Sc + kk + 16 + half * 8 + e];
    }
#pragma unroll
    for (int nt = 0; nt < 4; ++nt) {
      // Vt row (dk = nt*16 + r) is contiguous in s: one 32B fragment load
      v16bf bv = load_b_frag(&Vtb[(size_t)(nt * 16 + r) * Sc + kk], half);
      accv[nt] = WMMA_BF16(a, bv, accv[nt]);
    }
  }
#pragma unroll
  for (int nt = 0; nt < 4; ++nt)
#pragma unroll
    for (int i = 0; i < 8; ++i)
      part[wave][(half * 8 + i) * DKc + nt * 16 + r] = accv[nt][i];
  __syncthreads();

  // cross-wave reduction + store flat [B,S,D] bf16 for the out-projection
  for (int j = t; j < 16 * DKc; j += 256) {
    float s = 0.f;
#pragma unroll
    for (int w = 0; w < 8; ++w) s += part[w][j];
    const int row = j >> 6, col = j & 63;
    Obuf[((size_t)(b_ * Sc + qs + row)) * Dc + h_ * DKc + col] = (bf16_t)s;
  }
}

// ---------------------------------------------------------------------------
// Host launcher
// ---------------------------------------------------------------------------
extern "C" void kernel_launch(void* const* d_in, const int* in_sizes, int n_in,
                              void* d_out, int out_size, void* d_ws, size_t ws_size,
                              hipStream_t stream) {
  const float* q  = (const float*)d_in[0];
  const float* k  = (const float*)d_in[1];
  const float* v  = (const float*)d_in[2];
  const float* wq = (const float*)d_in[3];
  const float* bq = (const float*)d_in[4];
  const float* wk = (const float*)d_in[5];
  const float* bk = (const float*)d_in[6];
  const float* wv = (const float*)d_in[7];
  const float* bv = (const float*)d_in[8];
  const float* wo = (const float*)d_in[9];
  const float* bo = (const float*)d_in[10];

  float* out  = (float*)d_out;                       // [B,S,D]
  float* attn = out + (size_t)Bc * Sc * Dc;          // [B,H,S,S]

  // Workspace layout (64 MB total)
  char* ws = (char*)d_ws;
  const size_t actB = (size_t)Mc * Dc * sizeof(bf16_t);  // 8 MB
  const size_t wB   = (size_t)Dc * Dc * sizeof(bf16_t);  // 2 MB
  bf16_t* qb   = (bf16_t*)(ws);
  bf16_t* kb   = (bf16_t*)(ws + actB);
  bf16_t* vb   = (bf16_t*)(ws + 2 * actB);
  bf16_t* wqb  = (bf16_t*)(ws + 3 * actB);
  bf16_t* wkb  = (bf16_t*)(ws + 3 * actB + wB);
  bf16_t* wvb  = (bf16_t*)(ws + 3 * actB + 2 * wB);
  bf16_t* wob  = (bf16_t*)(ws + 3 * actB + 3 * wB);
  bf16_t* Qh   = (bf16_t*)(ws + 3 * actB + 4 * wB);
  bf16_t* Kh   = (bf16_t*)(ws + 4 * actB + 4 * wB);
  bf16_t* Vt   = (bf16_t*)(ws + 5 * actB + 4 * wB);   // [B,H,DK,S]
  bf16_t* Obuf = (bf16_t*)(ws + 6 * actB + 4 * wB);

  const int nAct = Mc * Dc;       // 4,194,304
  const int nW   = Dc * Dc;       // 1,048,576
  cast_f32_bf16<<<nAct / 1024, 256, 0, stream>>>(q,  qb,  nAct);
  cast_f32_bf16<<<nAct / 1024, 256, 0, stream>>>(k,  kb,  nAct);
  cast_f32_bf16<<<nAct / 1024, 256, 0, stream>>>(v,  vb,  nAct);
  cast_f32_bf16<<<nW   / 1024, 256, 0, stream>>>(wq, wqb, nW);
  cast_f32_bf16<<<nW   / 1024, 256, 0, stream>>>(wk, wkb, nW);
  cast_f32_bf16<<<nW   / 1024, 256, 0, stream>>>(wv, wvb, nW);
  cast_f32_bf16<<<nW   / 1024, 256, 0, stream>>>(wo, wob, nW);

  dim3 gg(Mc / 128, Dc / 128);    // 32 x 8
  gemm_bf16<<<gg, 256, 0, stream>>>(qb, wqb, bq, Qh, 0);
  gemm_bf16<<<gg, 256, 0, stream>>>(kb, wkb, bk, Kh, 0);
  gemm_bf16<<<gg, 256, 0, stream>>>(vb, wvb, bv, Vt, 2);

  dim3 ga(Sc / 16, Bc * Hc);      // 128 x 32
  attn_fused<<<ga, 256, 0, stream>>>(Qh, Kh, Vt, attn, Obuf);

  gemm_bf16<<<gg, 256, 0, stream>>>(Obuf, wob, bo, out, 1);
}
